// CrystalGraphE3ConvNet_17806934409756
// MI455X (gfx1250) — matrix operand order repaired and checked by reading, not compiled, not run on hardware
//
#include <hip/hip_runtime.h>

// ---------------------------------------------------------------------------
// CrystalGraphE3ConvNet for MI455X (gfx1250, wave32, WMMA bf16)
// ---------------------------------------------------------------------------

#define ATOM_F 64
#define NBR_F  41
#define H_F    128
#define N_CRYS 2000
#define Y0_CONST 0.28209479177387814f

typedef __attribute__((ext_vector_type(16))) __bf16 v16bf;
typedef __attribute__((ext_vector_type(8)))  __bf16 v8bf;
typedef __attribute__((ext_vector_type(8)))  float  v8f;

union bf_frag { v16bf v; v8bf h[2]; };

__device__ __forceinline__ float softplus_f(float x) {
    return (x > 20.0f) ? x : log1pf(__expf(x));
}

// ----------------------------- embedding -----------------------------------
__global__ __launch_bounds__(ATOM_F)
void cgc_embed_kernel(const float* __restrict__ atom_fea,
                      const float* __restrict__ emb_W,
                      const float* __restrict__ emb_b,
                      float* __restrict__ x, int N) {
    const int row = blockIdx.x;
    const int t = threadIdx.x;
    __shared__ float srow[92];
    for (int i = t; i < 92; i += ATOM_F) srow[i] = atom_fea[(size_t)row * 92 + i];
    __syncthreads();
    float acc = emb_b[t];
    #pragma unroll 4
    for (int k = 0; k < 92; ++k) acc += srow[k] * emb_W[k * ATOM_F + t];
    x[(size_t)row * ATOM_F + t] = acc;
}

// ----------------------------- neighbor counts -----------------------------
__global__ __launch_bounds__(256)
void cgc_counts_kernel(const int* __restrict__ nbr_idx,
                       float* __restrict__ counts, int NM) {
    int i = blockIdx.x * 256 + threadIdx.x;
    if (i < NM) atomicAdd(&counts[nbr_idx[i]], 1.0f);
}

// ----------------------------- conv layer ----------------------------------
// Per block: 128 edges. Gate MLP in fp32 VALU, 64x64 transform in bf16 WMMA,
// scatter via straight-line global_atomic_add_f32 (validity folded into scale).
__global__ __launch_bounds__(256)
void cgc_conv_kernel(const float* __restrict__ x_in,
                     float* __restrict__ summed,
                     const float* __restrict__ nbr_fea,
                     const int*   __restrict__ nbr_idx,
                     const float* __restrict__ rW1_l,   // [41,41]
                     const float* __restrict__ rb1_l,   // [41]
                     const float* __restrict__ rW2_l,   // [41,9] (col 0 used)
                     const float* __restrict__ rb2_l,   // [9]    ([0] used)
                     const float* __restrict__ tpW_l,   // [64,64]
                     int NM) {
    const int t = threadIdx.x;
    const int wave = t >> 5;
    const int lane = t & 31;
    const int edge0 = blockIdx.x * 128;

    __shared__ float  sh_rW1[NBR_F * NBR_F];
    __shared__ float  sh_rb1[NBR_F];
    __shared__ float  sh_w2c0[NBR_F];
    __shared__ float  sh_rad[128 * NBR_F];
    __shared__ float  sh_gpart[256];
    __shared__ float  sh_gate[128];
    __shared__ int    sh_nbr[128];
    __shared__ __align__(16) __bf16 sh_A[128 * ATOM_F];    // gated neigh feats
    __shared__ __align__(16) __bf16 sh_Bc[ATOM_F * ATOM_F]; // tpW COLUMN-major [n][k]

    // ---- stage weights / radial / neighbor indices ----
    for (int i = t; i < NBR_F * NBR_F; i += 256) sh_rW1[i] = rW1_l[i];
    for (int i = t; i < NBR_F; i += 256) {
        sh_rb1[i]  = rb1_l[i];
        sh_w2c0[i] = rW2_l[i * 9];           // column 0 of rW2
    }
    // tpW staged column-major so a B fragment is K-contiguous (ds_load_b128)
    for (int i = t; i < ATOM_F * ATOM_F; i += 256) {
        int n = i >> 6, k = i & 63;
        sh_Bc[n * ATOM_F + k] = (__bf16)tpW_l[k * ATOM_F + n];
    }
    for (int i = t; i < 128 * NBR_F; i += 256) {
        int e = i / NBR_F, k = i - e * NBR_F;
        int ge = edge0 + e;
        sh_rad[i] = (ge < NM) ? nbr_fea[(size_t)ge * NBR_F + k] : 0.0f;
    }
    if (t < 128) {
        int ge = edge0 + t;
        sh_nbr[t] = (ge < NM) ? nbr_idx[ge] : 0;
    }
    __syncthreads();

    // overlap the gather latency with the gate MLP (global_prefetch_b8)
    if (t < 128) __builtin_prefetch(&x_in[(size_t)sh_nbr[t] * ATOM_F], 0, 3);

    // ---- gate MLP: 2 threads per edge split the 41 hidden units ----
    {
        const int e = t >> 1;
        const int half = t & 1;
        const int j0 = half * 21;
        const int jn = half ? 20 : 21;
        const float* rad = &sh_rad[e * NBR_F];
        float p = 0.0f;
        for (int jj = 0; jj < jn; ++jj) {
            int j = j0 + jj;
            float z = sh_rb1[j];
            #pragma unroll 4
            for (int k = 0; k < NBR_F; ++k) z += rad[k] * sh_rW1[k * NBR_F + j];
            p += softplus_f(z) * sh_w2c0[j];
        }
        sh_gpart[t] = p;
    }
    __syncthreads();
    if (t < 128) {
        sh_gate[t] = Y0_CONST * (rb2_l[0] + sh_gpart[2 * t] + sh_gpart[2 * t + 1]);
    }
    __syncthreads();

    // ---- gather x[nbr], scale by gate, convert to bf16 A tile ----
    for (int i = t; i < 128 * ATOM_F; i += 256) {
        int e = i >> 6, f = i & 63;
        int ge = edge0 + e;
        float v = 0.0f;
        if (ge < NM) v = x_in[(size_t)sh_nbr[e] * ATOM_F + f] * sh_gate[e];
        sh_A[i] = (__bf16)v;
    }
    __syncthreads();

    // ---- WMMA: D[16,64] = A[16,64] @ B[64,64], per wave tile of 16 edges ---
    const __bf16* sAw = &sh_A[wave * 16 * ATOM_F];
    const int m_lane = lane & 15;
    const int hi = lane >> 4;

    // A fragments: two contiguous 16B runs per lane (ISA 16-bit A 16x32 layout)
    bf_frag af[2];
    #pragma unroll
    for (int kb = 0; kb < 2; ++kb) {
        const int kbase = hi * 8 + kb * 32;
        af[kb].h[0] = *(const v8bf*)(sAw + m_lane * ATOM_F + kbase);
        af[kb].h[1] = *(const v8bf*)(sAw + m_lane * ATOM_F + kbase + 16);
    }

    // per-lane scatter offsets + validity folded into the 1/sqrt(64) scale
    int   dofs[8];
    float dscale[8];
    #pragma unroll
    for (int r = 0; r < 8; ++r) {
        int m = r + 8 * hi;
        int ge = edge0 + wave * 16 + m;
        dofs[r]   = sh_nbr[wave * 16 + m] * ATOM_F + m_lane;
        dscale[r] = (ge < NM) ? 0.125f : 0.0f;
    }

    #pragma unroll
    for (int nt = 0; nt < 4; ++nt) {
        v8f acc = {};
        const int n = nt * 16 + m_lane;
        #pragma unroll
        for (int kb = 0; kb < 2; ++kb) {
            // B fragment: two K-contiguous 16B runs (k = 8*hi+s, 16+8*hi+s), matching A
            bf_frag bfr;
            const int kbb = hi * 8 + kb * 32;
            bfr.h[0] = *(const v8bf*)(sh_Bc + n * ATOM_F + kbb);
            bfr.h[1] = *(const v8bf*)(sh_Bc + n * ATOM_F + kbb + 16);
            acc = __builtin_amdgcn_wmma_f32_16x16x32_bf16(
                false, af[kb].v, false, bfr.v, (short)0, acc, false, false);
        }
        // straight-line scatter-add (C layout: VGPR r -> M = r + 8*hi, N = lane&15)
        #pragma unroll
        for (int r = 0; r < 8; ++r) {
            atomicAdd(&summed[dofs[r] + nt * 16], acc[r] * dscale[r]);
        }
    }
}

// ----------------------------- divide by counts ----------------------------
__global__ __launch_bounds__(256)
void cgc_divide_kernel(float* __restrict__ x, const float* __restrict__ counts,
                       int total) {
    int i = blockIdx.x * 256 + threadIdx.x;
    if (i < total) x[i] = x[i] / fmaxf(counts[i >> 6], 1.0f);
}

// ----------------------------- crystal pooling -----------------------------
__global__ __launch_bounds__(256)
void cgc_pool_kernel(const float* __restrict__ x,
                     const int* __restrict__ crys_idx,
                     float* __restrict__ csum, float* __restrict__ ccnt,
                     int total) {
    int i = blockIdx.x * 256 + threadIdx.x;
    if (i < total) {
        int n = i >> 6, f = i & 63;
        int c = crys_idx[n];
        if ((unsigned)c < (unsigned)N_CRYS) {   // XLA drops OOB scatter indices
            atomicAdd(&csum[(size_t)c * ATOM_F + f], x[i]);
            if (f == 0) atomicAdd(&ccnt[c], 1.0f);
        }
    }
}

// ----------------------------- head: fc + out ------------------------------
__global__ __launch_bounds__(H_F)
void cgc_head_kernel(const float* __restrict__ csum,
                     const float* __restrict__ ccnt,
                     const float* __restrict__ fc_W,   // [64,128]
                     const float* __restrict__ fc_b,   // [128]
                     const float* __restrict__ out_W,  // [128,1]
                     const float* __restrict__ out_b,  // [1]
                     float* __restrict__ out,          // [2000]
                     float* __restrict__ h_out) {      // [2000,128]
    const int c = blockIdx.x;
    const int t = threadIdx.x;
    __shared__ float scrys[ATOM_F];
    __shared__ float red[H_F];
    if (t < ATOM_F)
        scrys[t] = csum[(size_t)c * ATOM_F + t] / fmaxf(ccnt[c], 1.0f);
    __syncthreads();
    float acc = fc_b[t];
    #pragma unroll 4
    for (int k = 0; k < ATOM_F; ++k) acc += scrys[k] * fc_W[k * H_F + t];
    float hv = softplus_f(acc);
    h_out[(size_t)c * H_F + t] = hv;
    red[t] = hv * out_W[t];
    __syncthreads();
    for (int s = H_F / 2; s > 0; s >>= 1) {
        if (t < s) red[t] += red[t + s];
        __syncthreads();
    }
    if (t == 0) out[c] = red[0] + out_b[0];
}

// ---------------------------------------------------------------------------
extern "C" void kernel_launch(void* const* d_in, const int* in_sizes, int n_in,
                              void* d_out, int out_size, void* d_ws, size_t ws_size,
                              hipStream_t stream) {
    const float* atom_fea = (const float*)d_in[0];
    const float* nbr_fea  = (const float*)d_in[1];
    const int*   nbr_idx  = (const int*)d_in[2];
    const int*   crys_idx = (const int*)d_in[3];
    // d_in[4] = pos (unused by the reference computation)
    const float* emb_W = (const float*)d_in[5];
    const float* emb_b = (const float*)d_in[6];
    const float* rW1   = (const float*)d_in[7];
    const float* rb1   = (const float*)d_in[8];
    const float* rW2   = (const float*)d_in[9];
    const float* rb2   = (const float*)d_in[10];
    const float* tpW   = (const float*)d_in[11];
    const float* fc_W  = (const float*)d_in[12];
    const float* fc_b  = (const float*)d_in[13];
    const float* out_W = (const float*)d_in[14];
    const float* out_b = (const float*)d_in[15];

    const int N  = in_sizes[0] / 92;
    const int M  = in_sizes[2] / N;
    const int NM = N * M;

    // workspace carve-out
    float* bufA   = (float*)d_ws;
    float* bufB   = bufA + (size_t)N * ATOM_F;
    float* counts = bufB + (size_t)N * ATOM_F;
    float* csum   = counts + N;
    float* ccnt   = csum + (size_t)N_CRYS * ATOM_F;

    // embedding
    cgc_embed_kernel<<<N, ATOM_F, 0, stream>>>(atom_fea, emb_W, emb_b, bufA, N);

    // neighbor counts (layer-invariant)
    hipMemsetAsync(counts, 0, (size_t)N * sizeof(float), stream);
    cgc_counts_kernel<<<(NM + 255) / 256, 256, 0, stream>>>(nbr_idx, counts, NM);

    float* bufs[2] = {bufA, bufB};
    int cur = 0;
    for (int l = 0; l < 3; ++l) {
        float* xin  = bufs[cur];
        float* xout = bufs[cur ^ 1];
        hipMemsetAsync(xout, 0, (size_t)N * ATOM_F * sizeof(float), stream);
        cgc_conv_kernel<<<(NM + 127) / 128, 256, 0, stream>>>(
            xin, xout, nbr_fea, nbr_idx,
            rW1 + (size_t)l * NBR_F * NBR_F,
            rb1 + (size_t)l * NBR_F,
            rW2 + (size_t)l * NBR_F * 9,
            rb2 + (size_t)l * 9,
            tpW + (size_t)l * ATOM_F * ATOM_F,
            NM);
        cgc_divide_kernel<<<(N * ATOM_F + 255) / 256, 256, 0, stream>>>(
            xout, counts, N * ATOM_F);
        cur ^= 1;
    }
    const float* xfinal = bufs[cur];

    // per-crystal mean pooling
    hipMemsetAsync(csum, 0, (size_t)N_CRYS * ATOM_F * sizeof(float), stream);
    hipMemsetAsync(ccnt, 0, (size_t)N_CRYS * sizeof(float), stream);
    cgc_pool_kernel<<<(N * ATOM_F + 255) / 256, 256, 0, stream>>>(
        xfinal, crys_idx, csum, ccnt, N * ATOM_F);

    // head: out (2000) then h (2000 x 128), concatenated in return order
    float* out   = (float*)d_out;
    float* h_out = out + N_CRYS;
    cgc_head_kernel<<<N_CRYS, H_F, 0, stream>>>(csum, ccnt, fc_W, fc_b,
                                                out_W, out_b, out, h_out);
}
